// GCNTransformerGAT_35510789603747
// MI455X (gfx1250) — compile-verified, matrix operand order checked by reading, not run on hardware
//
#include <hip/hip_runtime.h>
#include <math.h>

typedef __attribute__((ext_vector_type(16))) _Float16 v16h;
typedef __attribute__((ext_vector_type(8)))  _Float16 v8h;
typedef __attribute__((ext_vector_type(8)))  float    v8f;

#define NN      20000
#define NE      320000
#define E2T     (NE + NN)      // edges + self loops
#define NH      4
#define CH      128            // per-head channels / HID
#define DM      512            // model dim (H*CH)
#define LRELU   0.2f

__device__ __forceinline__ v16h combine8(v8h lo, v8h hi)
{
    return __builtin_shufflevector(lo, hi, 0, 1, 2, 3, 4, 5, 6, 7,
                                           8, 9, 10, 11, 12, 13, 14, 15);
}

// ---------------------------------------------------------------------------
// Precision / layout conversion pre-passes (streaming, trivial vs GEMM cost)
// ---------------------------------------------------------------------------
__global__ void f32_to_f16(const float* __restrict__ in, _Float16* __restrict__ out, long n)
{
    long i = blockIdx.x * (long)blockDim.x + threadIdx.x;
    if (i < n) out[i] = (_Float16)in[i];
}

// in: R x C fp32 row-major;  out: C x R f16 (transposed)
__global__ void transpose_f32_to_f16(const float* __restrict__ in, _Float16* __restrict__ out,
                                     int R, int C)
{
    long i = blockIdx.x * (long)blockDim.x + threadIdx.x;
    if (i >= (long)R * C) return;
    const int r = (int)(i / C), c = (int)(i % C);
    out[(long)c * R + r] = (_Float16)in[i];
}

// X: S x DM fp32 -> out: [NH][S][CH] f16
__global__ void split_qk_f16(const float* __restrict__ X, _Float16* __restrict__ out, int S)
{
    long i = blockIdx.x * (long)blockDim.x + threadIdx.x;
    if (i >= (long)S * DM) return;
    const int s = (int)(i / DM), c = (int)(i % DM);
    const int h = c / CH, d = c % CH;
    out[((long)h * S + s) * CH + d] = (_Float16)X[i];
}

// X: S x DM fp32 -> out: [NH][CH][S] f16 (V transposed: key contiguous)
__global__ void split_vT_f16(const float* __restrict__ X, _Float16* __restrict__ out, int S)
{
    long i = blockIdx.x * (long)blockDim.x + threadIdx.x;
    if (i >= (long)S * DM) return;
    const int s = (int)(i / DM), c = (int)(i % DM);
    const int h = c / CH, d = c % CH;
    out[((long)h * CH + d) * S + s] = (_Float16)X[i];
}

// ---------------------------------------------------------------------------
// WMMA GEMM: C[M x Nc] = act(A16[M x K] @ BT16^T + bias), f16 in, f32 out.
// A16: M x K f16 row-major; BT16: Nc x K f16 (B transposed).
// Block = 8 waves covering 128 rows x 64 cols: all waves share the SAME
// 64-column B fragments (deduped in cache) and stream unique A rows.
// K-loop unrolled x2 into an explicit ping-pong (K % 64 == 0 at all call
// sites) so loop-carried fragments need no register copies: each load set
// is redefined only after its last WMMA use.
// All fragment loads are contiguous 16-byte b128 loads. M%16==0.
// ---------------------------------------------------------------------------
__global__ __launch_bounds__(256)
void gemm_wmma(const _Float16* __restrict__ A16, const _Float16* __restrict__ BT,
               const float* __restrict__ bias, float* __restrict__ C,
               int M, int K, int Nc, int act)
{
    const int lane  = threadIdx.x & 31;
    const int wave  = threadIdx.x >> 5;
    const int m0    = blockIdx.x * 128 + wave * 16;
    const int n0    = blockIdx.y * 64;
    if (m0 >= M || n0 >= Nc) return;

    const int half8 = (lane < 16) ? 0 : 8;
    const int l15   = lane & 15;
    const int row   = m0 + l15;              // valid: m0 < M and M % 16 == 0

    int  col[4], colc[4];
    bool cok[4];
#pragma unroll
    for (int t = 0; t < 4; ++t) {
        col[t]  = n0 + t * 16 + l15;
        cok[t]  = col[t] < Nc;
        colc[t] = cok[t] ? col[t] : 0;       // clamped: loads always legal
    }

    v8f acc[4];
#pragma unroll
    for (int t = 0; t < 4; ++t) acc[t] = (v8f){};

    const _Float16* Arow = A16 + (long)row * K + half8;
    const _Float16* Brow[4];
#pragma unroll
    for (int t = 0; t < 4; ++t) Brow[t] = BT + (long)colc[t] * K + half8;

    // ping-pong fragment sets
    v16h a0 = combine8(*(const v8h*)(Arow), *(const v8h*)(Arow + 16));
    v16h b0[4];
#pragma unroll
    for (int t = 0; t < 4; ++t)
        b0[t] = combine8(*(const v8h*)(Brow[t]), *(const v8h*)(Brow[t] + 16));

    for (int k0 = 0; k0 < K; k0 += 64) {
        const int k1 = k0 + 32;                       // always < K (K % 64 == 0)
        __builtin_prefetch(Arow + k1 + 32, 0, 1);
        const v16h a1 = combine8(*(const v8h*)(Arow + k1),
                                 *(const v8h*)(Arow + k1 + 16));
        v16h b1[4];
#pragma unroll
        for (int t = 0; t < 4; ++t)
            b1[t] = combine8(*(const v8h*)(Brow[t] + k1),
                             *(const v8h*)(Brow[t] + k1 + 16));
#pragma unroll
        for (int t = 0; t < 4; ++t)
            acc[t] = __builtin_amdgcn_wmma_f32_16x16x32_f16(false, a0, false, b0[t],
                                                            (short)0, acc[t], false, false);
        // set0 dead here -> reload in place (dummy on final iteration)
        const int k2 = (k1 + 32 < K) ? (k1 + 32) : 0;
        a0 = combine8(*(const v8h*)(Arow + k2), *(const v8h*)(Arow + k2 + 16));
#pragma unroll
        for (int t = 0; t < 4; ++t)
            b0[t] = combine8(*(const v8h*)(Brow[t] + k2),
                             *(const v8h*)(Brow[t] + k2 + 16));
#pragma unroll
        for (int t = 0; t < 4; ++t)
            acc[t] = __builtin_amdgcn_wmma_f32_16x16x32_f16(false, a1, false, b1[t],
                                                            (short)0, acc[t], false, false);
    }

#pragma unroll
    for (int t = 0; t < 4; ++t) {
        const float bv = bias ? bias[colc[t]] : 0.f;
#pragma unroll
        for (int r = 0; r < 8; ++r) {
            const int rr = m0 + r + half8;
            if (cok[t]) {
                float v = acc[t][r] + bv;
                if (act) v = fmaxf(v, 0.f);
                C[(long)rr * Nc + col[t]] = v;
            }
        }
    }
}

// ---------------------------------------------------------------------------
// Flash attention (S=20000, 4 heads, head dim 128). One wave per
// (16-query tile, head). f16 inputs in fragment-friendly layouts:
//   Qh,Kh: [NH][S][CH], VT: [NH][CH][S]. Online softmax over 32-key tiles;
// P converted C-layout -> A-layout through LDS; 16 WMMAs per key tile.
// ---------------------------------------------------------------------------
__global__ __launch_bounds__(32)
void attn_flash(const _Float16* __restrict__ Qh, const _Float16* __restrict__ Kh,
                const _Float16* __restrict__ VT, float* __restrict__ O, int S)
{
    __shared__ _Float16 psh[16 * 32];
    const int lane  = threadIdx.x;
    const int h     = blockIdx.y;
    const int q0    = blockIdx.x * 16;
    const int half8 = (lane < 16) ? 0 : 8;
    const int l15   = lane & 15;
    const float sc  = 0.088388347648318447f;   // 1/sqrt(128)

    // Q fragments: 16x128 as four 16x32 A-fragments (b128 loads)
    const _Float16* qrow = Qh + ((long)h * S + q0 + l15) * CH + half8;
    v16h qf[4];
#pragma unroll
    for (int f = 0; f < 4; ++f)
        qf[f] = combine8(*(const v8h*)(qrow + f * 32),
                         *(const v8h*)(qrow + f * 32 + 16));

    v8f o[8];
#pragma unroll
    for (int j = 0; j < 8; ++j) o[j] = (v8f){};
    float mr[8], lr[8];
#pragma unroll
    for (int r = 0; r < 8; ++r) { mr[r] = -1e30f; lr[r] = 0.f; }

    for (int kt = 0; kt < S; kt += 32) {
        // S = Q K^T for 32 keys -> two 16x16 C tiles
        v8f s0 = {}, s1 = {};
        const _Float16* k0r = Kh + ((long)h * S + kt + l15) * CH + half8;
        const _Float16* k1r = k0r + 16 * CH;
#pragma unroll
        for (int f = 0; f < 4; ++f) {
            const v16h kb0 = combine8(*(const v8h*)(k0r + f * 32),
                                      *(const v8h*)(k0r + f * 32 + 16));
            const v16h kb1 = combine8(*(const v8h*)(k1r + f * 32),
                                      *(const v8h*)(k1r + f * 32 + 16));
            s0 = __builtin_amdgcn_wmma_f32_16x16x32_f16(false, qf[f], false, kb0,
                                                        (short)0, s0, false, false);
            s1 = __builtin_amdgcn_wmma_f32_16x16x32_f16(false, qf[f], false, kb1,
                                                        (short)0, s1, false, false);
        }
        // online softmax; C-layout: value r lives at row (r + half8), col l15
        float p0[8], p1[8];
#pragma unroll
        for (int r = 0; r < 8; ++r) {
            const float v0 = s0[r] * sc, v1 = s1[r] * sc;
            float mx = fmaxf(v0, v1);
#pragma unroll
            for (int msk = 1; msk < 16; msk <<= 1)
                mx = fmaxf(mx, __shfl_xor(mx, msk, 32));
            const float mnew  = fmaxf(mr[r], mx);
            const float scale = __expf(mr[r] - mnew);
            p0[r] = __expf(v0 - mnew);
            p1[r] = __expf(v1 - mnew);
            float rs = p0[r] + p1[r];
#pragma unroll
            for (int msk = 1; msk < 16; msk <<= 1)
                rs += __shfl_xor(rs, msk, 32);
            lr[r] = lr[r] * scale + rs;
            mr[r] = mnew;
#pragma unroll
            for (int j = 0; j < 8; ++j) o[j][r] *= scale;
        }
        // C-layout -> A-layout via LDS (16x32 f16 P tile)
#pragma unroll
        for (int r = 0; r < 8; ++r) {
            const int rr = r + half8;
            psh[rr * 32 + l15]      = (_Float16)p0[r];
            psh[rr * 32 + 16 + l15] = (_Float16)p1[r];
        }
        __syncthreads();
        const v16h pa = combine8(*(const v8h*)(psh + l15 * 32 + half8),
                                 *(const v8h*)(psh + l15 * 32 + 16 + half8));
        __syncthreads();
        // O += P (16x32) x V (32 x 128), as eight 16x16 column blocks
#pragma unroll
        for (int j = 0; j < 8; ++j) {
            const _Float16* vrow = VT + ((long)h * CH + j * 16 + l15) * S + kt + half8;
            const v16h vb = combine8(*(const v8h*)(vrow),
                                     *(const v8h*)(vrow + 16));
            o[j] = __builtin_amdgcn_wmma_f32_16x16x32_f16(false, pa, false, vb,
                                                          (short)0, o[j], false, false);
        }
    }
#pragma unroll
    for (int j = 0; j < 8; ++j)
#pragma unroll
        for (int r = 0; r < 8; ++r) {
            const int rr = q0 + r + half8;
            O[(long)rr * DM + h * CH + j * 16 + l15] = o[j][r] / lr[r];
        }
}

// ---------------------------------------------------------------------------
// Graph / elementwise kernels
// ---------------------------------------------------------------------------
__device__ __forceinline__ void edge_sd(const int* ei, int e, int& s, int& d)
{
    if (e < NE) { s = ei[e]; d = ei[NE + e]; }
    else        { s = e - NE; d = s; }          // self loops
}

__global__ void fill_f(float* p, float v, long n)
{
    long i = blockIdx.x * (long)blockDim.x + threadIdx.x;
    if (i < n) p[i] = v;
}

__global__ void node_dots(const float* __restrict__ h, const float* __restrict__ asrc,
                          const float* __restrict__ adst, float* as_, float* ad_)
{
    int i = blockIdx.x * blockDim.x + threadIdx.x;  // over N*H
    if (i >= NN * NH) return;
    const int n = i / NH, hh = i % NH;
    const float* hp = h + (long)n * DM + hh * CH;
    const float* sp = asrc + hh * CH;
    const float* dp = adst + hh * CH;
    float s = 0.f, d = 0.f;
    for (int c = 0; c < CH; ++c) { const float v = hp[c]; s += v * sp[c]; d += v * dp[c]; }
    as_[i] = s; ad_[i] = d;
}

__device__ __forceinline__ void atomicMaxF(float* addr, float val)
{
    unsigned* ua  = (unsigned*)addr;
    unsigned  old = *ua;
    while (true) {
        if (__uint_as_float(old) >= val) break;
        const unsigned assumed = old;
        old = atomicCAS(ua, assumed, __float_as_uint(val));
        if (old == assumed) break;
    }
}

__global__ void edge_logit_max(const int* ei, const float* as_, const float* ad_,
                               float* elog, float* segmax)
{
    long i = blockIdx.x * (long)blockDim.x + threadIdx.x;  // over E2T*H
    if (i >= (long)E2T * NH) return;
    const int e = (int)(i / NH), hh = (int)(i % NH);
    int s, d; edge_sd(ei, e, s, d);
    float v = as_[s * NH + hh] + ad_[d * NH + hh];
    v = (v > 0.f) ? v : LRELU * v;
    elog[i] = v;
    atomicMaxF(&segmax[d * NH + hh], v);
}

__global__ void edge_expsum(const int* ei, float* elog, const float* segmax, float* segsum)
{
    long i = blockIdx.x * (long)blockDim.x + threadIdx.x;
    if (i >= (long)E2T * NH) return;
    const int e = (int)(i / NH), hh = (int)(i % NH);
    int s, d; edge_sd(ei, e, s, d);
    const float ee = __expf(elog[i] - segmax[d * NH + hh]);
    elog[i] = ee;
    atomicAdd(&segsum[d * NH + hh], ee);
}

__global__ void edge_scatter_gat(const int* ei, const float* __restrict__ h,
                                 const float* elog, const float* segsum, float* out)
{
    long i = blockIdx.x * (long)blockDim.x + threadIdx.x;
    if (i >= (long)E2T * NH) return;
    const int e = (int)(i / NH), hh = (int)(i % NH);
    int s, d; edge_sd(ei, e, s, d);
    const float coef = elog[i] / (segsum[d * NH + hh] + 1e-16f);
    const float* hp = h + (long)s * DM + hh * CH;
    float* op = out + (long)d * DM + hh * CH;
    for (int c = 0; c < CH; ++c) atomicAdd(&op[c], hp[c] * coef);
}

__global__ void bias_act_k(float* x, const float* b, long M, int C, int act)
{
    long i = blockIdx.x * (long)blockDim.x + threadIdx.x;
    if (i >= M * C) return;
    const int c = (int)(i % C);
    float v = x[i] + b[c];
    if (act) v = fmaxf(v, 0.f);
    x[i] = v;
}

__global__ void bn_stats(const float* x, float* sum, float* sq, long M, int C)
{
    long i = blockIdx.x * (long)blockDim.x + threadIdx.x;
    if (i >= M * C) return;
    const int c = (int)(i % C);
    const float v = x[i];
    atomicAdd(&sum[c], v);
    atomicAdd(&sq[c], v * v);
}

__global__ void bn_apply(float* x, const float* sum, const float* sq,
                         const float* g, const float* b, long M, int C)
{
    long i = blockIdx.x * (long)blockDim.x + threadIdx.x;
    if (i >= M * C) return;
    const int c = (int)(i % C);
    const float mu  = sum[c] / (float)M;
    const float var = sq[c] / (float)M - mu * mu;
    x[i] = g[c] * (x[i] - mu) * rsqrtf(var + 1e-5f) + b[c];
}

__global__ void deg_k(const int* ei, const float* ew, float* deg)
{
    int e = blockIdx.x * blockDim.x + threadIdx.x;
    if (e >= E2T) return;
    int s, d; edge_sd(ei, e, s, d);
    const float w = (e < NE) ? ew[e] : 1.f;
    atomicAdd(&deg[d], w);
}

__global__ void dinv_k(float* deg)
{
    int n = blockIdx.x * blockDim.x + threadIdx.x;
    if (n >= NN) return;
    const float dg = deg[n];
    deg[n] = (dg > 0.f) ? rsqrtf(dg) : 0.f;
}

__global__ void edge_scatter_gcn(const int* ei, const float* ew, const float* dinv,
                                 const float* __restrict__ h, float* out, int C)
{
    int e = blockIdx.x * blockDim.x + threadIdx.x;
    if (e >= E2T) return;
    int s, d; edge_sd(ei, e, s, d);
    const float w    = (e < NE) ? ew[e] : 1.f;
    const float coef = dinv[s] * w * dinv[d];
    const float* hp = h + (long)s * C;
    float* op = out + (long)d * C;
    for (int c = 0; c < C; ++c) atomicAdd(&op[c], hp[c] * coef);
}

// x = LayerNorm(x + r); one wave per row of DM=512
__global__ __launch_bounds__(32)
void residual_ln(float* x, const float* r, const float* g, const float* b)
{
    const int row  = blockIdx.x;
    const int lane = threadIdx.x;
    float* xp = x + (long)row * DM;
    const float* rp = r + (long)row * DM;
    float vals[16];
    float s = 0.f;
#pragma unroll
    for (int i = 0; i < 16; ++i) {
        vals[i] = xp[lane + 32 * i] + rp[lane + 32 * i];
        s += vals[i];
    }
#pragma unroll
    for (int msk = 1; msk < 32; msk <<= 1) s += __shfl_xor(s, msk, 32);
    const float mu = s / (float)DM;
    float q = 0.f;
#pragma unroll
    for (int i = 0; i < 16; ++i) { const float dd = vals[i] - mu; q += dd * dd; }
#pragma unroll
    for (int msk = 1; msk < 32; msk <<= 1) q += __shfl_xor(q, msk, 32);
    const float inv = rsqrtf(q / (float)DM + 1e-5f);
#pragma unroll
    for (int i = 0; i < 16; ++i) {
        const int c = lane + 32 * i;
        xp[c] = g[c] * (vals[i] - mu) * inv + b[c];
    }
}

// ---------------------------------------------------------------------------
// Host driver
// ---------------------------------------------------------------------------
extern "C" void kernel_launch(void* const* d_in, const int* in_sizes, int n_in,
                              void* d_out, int out_size, void* d_ws, size_t ws_size,
                              hipStream_t stream)
{
    (void)in_sizes; (void)n_in; (void)out_size; (void)ws_size;
    const float* x_in = (const float*)d_in[0];
    const int*   ei   = (const int*)d_in[1];
    const float* ew   = (const float*)d_in[2];
    auto F = [&](int i) { return (const float*)d_in[i]; };

    char* wsb = (char*)d_ws;
    size_t off = 0;
    auto allocf = [&](size_t n) { float* p = (float*)(wsb + off);
                                  off = (off + n * 4 + 15) & ~(size_t)15; return p; };
    auto alloch = [&](size_t n) { _Float16* p = (_Float16*)(wsb + off);
                                  off = (off + n * 2 + 15) & ~(size_t)15; return p; };
    float* bufX  = allocf((size_t)NN * DM);
    float* bufH  = allocf((size_t)NN * DM);
    float* bufT  = allocf((size_t)NN * 2048);
    float* Qb    = allocf((size_t)NN * DM);
    float* Kb    = allocf((size_t)NN * DM);
    float* Vb    = allocf((size_t)NN * DM);
    float* asb   = allocf((size_t)NN * NH);
    float* adb   = allocf((size_t)NN * NH);
    float* elog  = allocf((size_t)E2T * NH);
    float* smax  = allocf((size_t)NN * NH);
    float* ssum  = allocf((size_t)NN * NH);
    float* degb  = allocf((size_t)NN);
    float* bnsum = allocf(2048);
    float* bnsq  = allocf(2048);
    _Float16* A16  = alloch((size_t)NN * 2048 + 64);
    _Float16* B16  = alloch((size_t)2048 * 512 + 64);
    _Float16* Qh16 = alloch((size_t)NH * NN * CH);
    _Float16* Kh16 = alloch((size_t)NH * NN * CH);
    _Float16* VT16 = alloch((size_t)NH * NN * CH);

    auto fill = [&](float* p, float v, long n) {
        fill_f<<<(unsigned)((n + 255) / 256), 256, 0, stream>>>(p, v, n);
    };
    auto gemm = [&](const float* A, const float* B, const float* bias, float* C,
                    int M, int K, int Nc, int act) {
        const long na = (long)M * K, nb = (long)K * Nc;
        f32_to_f16<<<(unsigned)((na + 255) / 256), 256, 0, stream>>>(A, A16, na);
        transpose_f32_to_f16<<<(unsigned)((nb + 255) / 256), 256, 0, stream>>>(B, B16, K, Nc);
        dim3 grid((M + 127) / 128, (Nc + 63) / 64);
        gemm_wmma<<<grid, 256, 0, stream>>>(A16, B16, bias, C, M, K, Nc, act);
    };
    const long nhE = (long)E2T * NH;
    auto run_gat = [&](const float* xin, int Fin, const float* W, const float* a_s,
                       const float* a_d, const float* bias, float* out) {
        gemm(xin, W, nullptr, bufH, NN, Fin, DM, 0);
        node_dots<<<(NN * NH + 255) / 256, 256, 0, stream>>>(bufH, a_s, a_d, asb, adb);
        fill(smax, -1e30f, (long)NN * NH);
        fill(ssum, 0.f, (long)NN * NH);
        edge_logit_max<<<(unsigned)((nhE + 255) / 256), 256, 0, stream>>>(ei, asb, adb, elog, smax);
        edge_expsum<<<(unsigned)((nhE + 255) / 256), 256, 0, stream>>>(ei, elog, smax, ssum);
        fill(out, 0.f, (long)NN * DM);
        edge_scatter_gat<<<(unsigned)((nhE + 255) / 256), 256, 0, stream>>>(ei, bufH, elog, ssum, out);
        bias_act_k<<<(unsigned)(((long)NN * DM + 255) / 256), 256, 0, stream>>>(out, bias, NN, DM, 1);
    };
    auto run_bn = [&](float* xbuf, int C, const float* g, const float* bb) {
        fill(bnsum, 0.f, C);
        fill(bnsq, 0.f, C);
        bn_stats<<<(unsigned)(((long)NN * C + 255) / 256), 256, 0, stream>>>(xbuf, bnsum, bnsq, NN, C);
        bn_apply<<<(unsigned)(((long)NN * C + 255) / 256), 256, 0, stream>>>(xbuf, bnsum, bnsq, g, bb, NN, C);
    };
    auto run_tf = [&](float* x, int base) {
        gemm(x, F(base + 0), F(base + 1), Qb, NN, DM, DM, 0);
        gemm(x, F(base + 2), F(base + 3), Kb, NN, DM, DM, 0);
        gemm(x, F(base + 4), F(base + 5), Vb, NN, DM, DM, 0);
        const long nsd = (long)NN * DM;
        split_qk_f16<<<(unsigned)((nsd + 255) / 256), 256, 0, stream>>>(Qb, Qh16, NN);
        split_qk_f16<<<(unsigned)((nsd + 255) / 256), 256, 0, stream>>>(Kb, Kh16, NN);
        split_vT_f16<<<(unsigned)((nsd + 255) / 256), 256, 0, stream>>>(Vb, VT16, NN);
        dim3 ag(NN / 16, NH);
        attn_flash<<<ag, 32, 0, stream>>>(Qh16, Kh16, VT16, bufH, NN);
        gemm(bufH, F(base + 6), F(base + 7), bufT, NN, DM, DM, 0);
        residual_ln<<<NN, 32, 0, stream>>>(x, bufT, F(base + 8), F(base + 9));
        gemm(x, F(base + 10), F(base + 11), bufT, NN, DM, 2048, 1);
        gemm(bufT, F(base + 12), F(base + 13), bufH, NN, 2048, DM, 0);
        residual_ln<<<NN, 32, 0, stream>>>(x, bufH, F(base + 14), F(base + 15));
    };

    // ---- GAT1 + BN ----
    run_gat(x_in, 256, F(3), F(4), F(5), F(6), bufX);
    run_bn(bufX, DM, F(7), F(8));

    // ---- GCN degree normalization ----
    fill(degb, 0.f, NN);
    deg_k<<<(E2T + 255) / 256, 256, 0, stream>>>(ei, ew, degb);
    dinv_k<<<(NN + 255) / 256, 256, 0, stream>>>(degb);

    // ---- 4 GCN layers ----
    float* cur = bufX;
    float* nxt = bufT;
    for (int i = 0; i < 4; ++i) {
        const int Cin = (i == 0) ? DM : CH;
        gemm(cur, F(9 + 2 * i), nullptr, bufH, NN, Cin, CH, 0);
        fill(nxt, 0.f, (long)NN * CH);
        edge_scatter_gcn<<<(E2T + 255) / 256, 256, 0, stream>>>(ei, ew, degb, bufH, nxt, CH);
        bias_act_k<<<(unsigned)(((long)NN * CH + 255) / 256), 256, 0, stream>>>(nxt, F(10 + 2 * i), NN, CH, 1);
        run_bn(nxt, CH, F(17 + 2 * i), F(18 + 2 * i));
        float* t = cur; cur = nxt; nxt = t;
    }
    // cur == bufX holds N x 128

    // ---- GAT2 + BN (into bufT) ----
    run_gat(cur, CH, F(25), F(26), F(27), F(28), bufT);
    run_bn(bufT, DM, F(29), F(30));

    // ---- proj ----
    gemm(bufT, F(31), F(32), bufX, NN, DM, DM, 0);

    // ---- 2 transformer layers ----
    run_tf(bufX, 33);
    run_tf(bufX, 49);

    // ---- head ----
    gemm(bufX, F(65), F(66), bufT, NN, DM, DM, 1);
    gemm(bufT, F(67), F(68), (float*)d_out, NN, DM, 40, 0);
}
